// CausalSelfAttention_2121713844495
// MI455X (gfx1250) — matrix operand order repaired and checked by reading, not compiled, not run on hardware
//
#include <hip/hip_runtime.h>

typedef __bf16 bf16;
typedef __attribute__((ext_vector_type(16))) __bf16 v16bf;
typedef __attribute__((ext_vector_type(8)))  float  v8f;

constexpr int Bsz = 8, Tsz = 1024, Csz = 768, NHn = 12, HDd = 64, C3 = 2304;

__device__ __forceinline__ v8f wmma_bf16(v16bf a, v16bf b, v8f c) {
  // D = A(16x32 bf16) * B(32x16 bf16) + C(16x16 f32)
  return __builtin_amdgcn_wmma_f32_16x16x32_bf16(false, a, false, b, (short)0, c,
                                                 false, false);
}

// A fragment, 16x32 tile, row-major source with leading dim lda.
// Contiguous k order (k = 16h + s, same as B loaders): lanes 0-15 -> row=lane, elems = K0..15 ; lanes 16-31 -> K16..31
__device__ __forceinline__ v16bf load_A_frag(const bf16* A, int lda, int lane) {
  const int r  = lane & 15;
  const int kh = (lane >> 4) << 4;  // 0 or 16
  const bf16* row = A + (size_t)r * lda;
  v16bf a;
#pragma unroll
  for (int i = 0; i < 8; ++i) {
    a[i]     = row[kh + i];
    a[i + 8] = row[8 + kh + i];
  }
  return a;
}

// B fragment from TRANSPOSED storage BT[n][k] (contiguous along k).
// Contiguous k order (same as load_A_frag): n = lane%16; lanes 0-15 hold K0..15, lanes 16-31 hold K16..31.
__device__ __forceinline__ v16bf load_B_fragT(const bf16* BT, int ldb, int lane) {
  const int n  = lane & 15;
  const int kb = (lane >> 4) << 4;  // 0 or 16
  const bf16* row = BT + (size_t)n * ldb + kb;
  v16bf b;
#pragma unroll
  for (int i = 0; i < 16; ++i) b[i] = row[i];
  return b;
}

// B fragment from natural row-major B[k][n] (strided along k).
__device__ __forceinline__ v16bf load_B_frag(const bf16* Bm, int ldb, int lane) {
  const int n  = lane & 15;
  const int kb = (lane >> 4) << 4;
  v16bf b;
#pragma unroll
  for (int i = 0; i < 16; ++i) b[i] = Bm[(size_t)(kb + i) * ldb + n];
  return b;
}

// ---------------- conversion kernels ----------------

__global__ void cvt_f32_bf16_kernel(const float* __restrict__ src,
                                    bf16* __restrict__ dst, int n) {
  int i = blockIdx.x * blockDim.x + threadIdx.x;
  if (i < n) dst[i] = (bf16)src[i];
}

// src[rows][cols] fp32 -> dst[cols][rows] bf16
__global__ void transpose_cvt_kernel(const float* __restrict__ src,
                                     bf16* __restrict__ dst, int rows, int cols) {
  int i = blockIdx.x * blockDim.x + threadIdx.x;
  if (i < rows * cols) {
    int r = i / cols, c = i % cols;
    dst[(size_t)c * rows + r] = (bf16)src[i];
  }
}

// ---------------- register-blocked GEMM core: 32(M) x 64(N) per wave ----------------
// A row-major [M][K], BT transposed weights [N][K]. 8 WMMAs per K-step of 32.
__device__ __forceinline__ void gemm_wave_32x64(const bf16* __restrict__ A, int lda,
                                                const bf16* __restrict__ BT, int ldb,
                                                int K, int lane, v8f acc[2][4]) {
  for (int kk = 0; kk < K; kk += 32) {
    const v16bf a0 = load_A_frag(A + kk, lda, lane);
    const v16bf a1 = load_A_frag(A + (size_t)16 * lda + kk, lda, lane);
#pragma unroll
    for (int na = 0; na < 4; ++na) {
      const v16bf bfr = load_B_fragT(BT + (size_t)(na * 16) * ldb + kk, ldb, lane);
      acc[0][na] = wmma_bf16(a0, bfr, acc[0][na]);
      acc[1][na] = wmma_bf16(a1, bfr, acc[1][na]);
    }
  }
}

// ---------------- QKV GEMM: [8192 x 768] @ [768 x 2304] ----------------
// grid (256, 9), block 128 (4 waves). Wave computes a 32x64 tile.
// Writes q (scaled by 1/sqrt(HD)), k, v in [B,NH,T,HD] bf16 layout.
__global__ void __launch_bounds__(128)
qkv_gemm_kernel(const bf16* __restrict__ xb, const bf16* __restrict__ waT,
                const float* __restrict__ bias, bf16* __restrict__ q,
                bf16* __restrict__ k, bf16* __restrict__ v) {
  const int lane = threadIdx.x & 31;
  const int wave = threadIdx.x >> 5;
  const int m0 = blockIdx.x * 32;
  const int n0 = blockIdx.y * 256 + wave * 64;   // 64-aligned -> one head per 16-chunk

  v8f acc[2][4] = {};
  gemm_wave_32x64(xb + (size_t)m0 * Csz, Csz, waT + (size_t)n0 * Csz, Csz, Csz, lane, acc);

  const int lane15 = lane & 15;
  const int rbase  = (lane >> 4) << 3;  // 0 or 8
#pragma unroll
  for (int na = 0; na < 4; ++na) {
    const int n = n0 + na * 16 + lane15;
    const float bval = bias[n];
    const int which = n / Csz;          // 0=q 1=k 2=v (constant over the 16-chunk)
    const int cidx  = n % Csz;
    const int h = cidx >> 6, d = cidx & 63;
    bf16* dst = (which == 0) ? q : ((which == 1) ? k : v);
    const float scale = (which == 0) ? 0.125f : 1.0f;  // 1/sqrt(64) folded into q
#pragma unroll
    for (int mi = 0; mi < 2; ++mi) {
#pragma unroll
      for (int j = 0; j < 8; ++j) {
        const int row = m0 + mi * 16 + rbase + j;  // global row = b*T + t
        const int bb = row >> 10, t = row & 1023;
        dst[(((size_t)(bb * NHn + h)) * Tsz + t) * HDd + d] =
            (bf16)((acc[mi][na][j] + bval) * scale);
      }
    }
  }
}

// ---------------- Flash attention ----------------
// Block = 4 waves = 64 query rows (uniform key loop -> safe __syncthreads).
// Per wave: 16 query rows, O accumulated over HD=64 as 4x v8f, online softmax.
__global__ void __launch_bounds__(128)
flash_attn_kernel(const bf16* __restrict__ q, const bf16* __restrict__ k,
                  const bf16* __restrict__ v, bf16* __restrict__ y) {
  __shared__ bf16 pbuf[4][16][36];  // per-wave P tile (16 q x 32 keys), padded

  const int lane   = threadIdx.x & 31;
  const int wave   = threadIdx.x >> 5;
  const int lane15 = lane & 15;
  const int rbase  = (lane >> 4) << 3;

  const int bid = blockIdx.x;
  const int q64 = bid & 15;            // 16 blocks of 64 q-rows per (b,h)
  const int h   = (bid >> 4) % NHn;
  const int b   = bid / (16 * NHn);
  const int q0blk = q64 * 64;
  const int q0w   = q0blk + wave * 16;

  const bf16* Qh = q + (size_t)(b * NHn + h) * Tsz * HDd;
  const bf16* Kh = k + (size_t)(b * NHn + h) * Tsz * HDd;
  const bf16* Vh = v + (size_t)(b * NHn + h) * Tsz * HDd;

  const v16bf qa0 = load_A_frag(Qh + (size_t)q0w * HDd + 0,  HDd, lane);
  const v16bf qa1 = load_A_frag(Qh + (size_t)q0w * HDd + 32, HDd, lane);

  v8f o0 = {}, o1 = {}, o2 = {}, o3 = {};
  float mrun[8], lrun[8];
#pragma unroll
  for (int j = 0; j < 8; ++j) { mrun[j] = -3.0e38f; lrun[j] = 0.0f; }

  const int kend = q0blk + 64;  // keys up to block's last query, inclusive
  for (int kb = 0; kb < kend; kb += 32) {
    // S tiles: keys [kb, kb+16) and [kb+16, kb+32); K dim = HD via 2 WMMAs each.
    // K^T as B: B[d][n] = K[kb+n][d] -> transposed-storage loader on row-major K.
    v8f s0 = {}, s1 = {};
    s0 = wmma_bf16(qa0, load_B_fragT(Kh + (size_t)(kb)      * HDd + 0,  HDd, lane), s0);
    s0 = wmma_bf16(qa1, load_B_fragT(Kh + (size_t)(kb)      * HDd + 32, HDd, lane), s0);
    s1 = wmma_bf16(qa0, load_B_fragT(Kh + (size_t)(kb + 16) * HDd + 0,  HDd, lane), s1);
    s1 = wmma_bf16(qa1, load_B_fragT(Kh + (size_t)(kb + 16) * HDd + 32, HDd, lane), s1);

    const int key0 = kb + lane15;
    const int key1 = kb + 16 + lane15;
#pragma unroll
    for (int j = 0; j < 8; ++j) {
      const int qrow = q0w + rbase + j;
      float t0 = (key0 <= qrow) ? s0[j] : -3.0e38f;
      float t1 = (key1 <= qrow) ? s1[j] : -3.0e38f;
      float mx = fmaxf(t0, t1);
      mx = fmaxf(mx, __shfl_xor(mx, 1, 32));
      mx = fmaxf(mx, __shfl_xor(mx, 2, 32));
      mx = fmaxf(mx, __shfl_xor(mx, 4, 32));
      mx = fmaxf(mx, __shfl_xor(mx, 8, 32));
      const float mn = fmaxf(mrun[j], mx);
      float p0 = (key0 <= qrow) ? __expf(t0 - mn) : 0.0f;
      float p1 = (key1 <= qrow) ? __expf(t1 - mn) : 0.0f;
      float rs = p0 + p1;
      rs += __shfl_xor(rs, 1, 32);
      rs += __shfl_xor(rs, 2, 32);
      rs += __shfl_xor(rs, 4, 32);
      rs += __shfl_xor(rs, 8, 32);
      const float alpha = __expf(mrun[j] - mn);
      lrun[j] = lrun[j] * alpha + rs;
      mrun[j] = mn;
      o0[j] *= alpha; o1[j] *= alpha; o2[j] *= alpha; o3[j] *= alpha;
      pbuf[wave][rbase + j][lane15]      = (bf16)p0;   // C-layout -> LDS
      pbuf[wave][rbase + j][16 + lane15] = (bf16)p1;
    }
    __syncthreads();

    // Reload P in A-fragment layout (16 q x 32 keys).
    v16bf pa;
    {
      const bf16* prow = &pbuf[wave][lane15][0];
      const int kh = (lane >> 4) << 4;
#pragma unroll
      for (int i = 0; i < 8; ++i) {
        pa[i]     = prow[kh + i];
        pa[i + 8] = prow[8 + kh + i];
      }
    }
    // O(16 x HD) += P(16x32) * V(32 x HD); V row-major [T][HD].
    o0 = wmma_bf16(pa, load_B_frag(Vh + (size_t)kb * HDd + 0,  HDd, lane), o0);
    o1 = wmma_bf16(pa, load_B_frag(Vh + (size_t)kb * HDd + 16, HDd, lane), o1);
    o2 = wmma_bf16(pa, load_B_frag(Vh + (size_t)kb * HDd + 32, HDd, lane), o2);
    o3 = wmma_bf16(pa, load_B_frag(Vh + (size_t)kb * HDd + 48, HDd, lane), o3);
    __syncthreads();
  }

  // Normalize and store to y[B,T,C] (bf16) for the out-projection GEMM.
#pragma unroll
  for (int j = 0; j < 8; ++j) {
    const float invl = (lrun[j] > 0.0f) ? (1.0f / lrun[j]) : 0.0f;
    const int t = q0w + rbase + j;
    const size_t base = ((size_t)b * Tsz + t) * Csz + h * HDd;
    y[base + 0  + lane15] = (bf16)(o0[j] * invl);
    y[base + 16 + lane15] = (bf16)(o1[j] * invl);
    y[base + 32 + lane15] = (bf16)(o2[j] * invl);
    y[base + 48 + lane15] = (bf16)(o3[j] * invl);
  }
}

// ---------------- Out projection: [8192 x 768] @ [768 x 768] -> fp32 ----------------
// grid (256, 3), block 128 (4 waves). Wave computes a 32x64 tile.
__global__ void __launch_bounds__(128)
proj_gemm_kernel(const bf16* __restrict__ yb, const bf16* __restrict__ wpT,
                 const float* __restrict__ bias, float* __restrict__ out) {
  const int lane = threadIdx.x & 31;
  const int wave = threadIdx.x >> 5;
  const int m0 = blockIdx.x * 32;
  const int n0 = blockIdx.y * 256 + wave * 64;

  v8f acc[2][4] = {};
  gemm_wave_32x64(yb + (size_t)m0 * Csz, Csz, wpT + (size_t)n0 * Csz, Csz, Csz, lane, acc);

  const int lane15 = lane & 15;
  const int rbase  = (lane >> 4) << 3;
#pragma unroll
  for (int na = 0; na < 4; ++na) {
    const int n = n0 + na * 16 + lane15;
    const float bval = bias[n];
#pragma unroll
    for (int mi = 0; mi < 2; ++mi) {
#pragma unroll
      for (int j = 0; j < 8; ++j) {
        const int row = m0 + mi * 16 + rbase + j;
        out[(size_t)row * Csz + n] = acc[mi][na][j] + bval;
      }
    }
  }
}

// ---------------- host launch ----------------
extern "C" void kernel_launch(void* const* d_in, const int* in_sizes, int n_in,
                              void* d_out, int out_size, void* d_ws, size_t ws_size,
                              hipStream_t stream) {
  (void)in_sizes; (void)n_in; (void)out_size; (void)ws_size;
  const float* x      = (const float*)d_in[0];
  const float* w_attn = (const float*)d_in[1];
  const float* b_attn = (const float*)d_in[2];
  const float* w_proj = (const float*)d_in[3];
  const float* b_proj = (const float*)d_in[4];
  float* out = (float*)d_out;

  char* ws = (char*)d_ws;
  size_t off = 0;
  auto carve = [&](size_t bytes) -> char* {
    char* p = ws + off;
    off = (off + bytes + 255) & ~(size_t)255;
    return p;
  };
  bf16* xb   = (bf16*)carve((size_t)Bsz * Tsz * Csz * 2);
  bf16* waT  = (bf16*)carve((size_t)C3 * Csz * 2);
  bf16* wpT  = (bf16*)carve((size_t)Csz * Csz * 2);
  bf16* qb   = (bf16*)carve((size_t)Bsz * NHn * Tsz * HDd * 2);
  bf16* kbuf = (bf16*)carve((size_t)Bsz * NHn * Tsz * HDd * 2);
  bf16* vbuf = (bf16*)carve((size_t)Bsz * NHn * Tsz * HDd * 2);
  bf16* yb   = (bf16*)carve((size_t)Bsz * Tsz * Csz * 2);

  const int nX = Bsz * Tsz * Csz;
  cvt_f32_bf16_kernel<<<(nX + 255) / 256, 256, 0, stream>>>(x, xb, nX);
  transpose_cvt_kernel<<<(Csz * C3 + 255) / 256, 256, 0, stream>>>(w_attn, waT, Csz, C3);
  transpose_cvt_kernel<<<(Csz * Csz + 255) / 256, 256, 0, stream>>>(w_proj, wpT, Csz, Csz);

  qkv_gemm_kernel<<<dim3(Bsz * Tsz / 32, C3 / 256), 128, 0, stream>>>(xb, waT, b_attn,
                                                                      qb, kbuf, vbuf);
  flash_attn_kernel<<<Bsz * NHn * (Tsz / 64), 128, 0, stream>>>(qb, kbuf, vbuf, yb);
  proj_gemm_kernel<<<dim3(Bsz * Tsz / 32, Csz / 256), 128, 0, stream>>>(yb, wpT, b_proj, out);
}
